// ProximalAttention_12807592476905
// MI455X (gfx1250) — matrix operand order repaired and checked by reading, not compiled, not run on hardware
//
#include <hip/hip_runtime.h>
#include <cstdint>
#include <cstddef>

// Problem constants (from reference): B=4, I=A=2048, E=256, H=4, D=64
#define BATCH 4
#define SEQ   2048
#define EMB   256
#define HEADS 4
#define DHEAD 64
#define ATILE 64     // keys per flash-attention iteration

typedef _Float16 v16h __attribute__((ext_vector_type(16)));
typedef float    v8f  __attribute__((ext_vector_type(8)));
typedef _Float16 hf2  __attribute__((ext_vector_type(2)));
typedef int      v4i  __attribute__((ext_vector_type(4)));

typedef __attribute__((address_space(1))) v4i g_v4i;   // global 16B vector
typedef __attribute__((address_space(3))) v4i l_v4i;   // LDS 16B vector

union F16x16 { uint32_t u[8]; v16h v; };

#if __has_builtin(__builtin_amdgcn_global_load_async_to_lds_b128)
#define HAVE_ASYNC_LDS 1
#else
#define HAVE_ASYNC_LDS 0
#endif

#if HAVE_ASYNC_LDS
__device__ __forceinline__ void async_copy16(const _Float16* g, _Float16* lds)
{
    __builtin_amdgcn_global_load_async_to_lds_b128((g_v4i*)g, (l_v4i*)lds, 0, 0);
}
__device__ __forceinline__ void wait_async_zero()
{
#if __has_builtin(__builtin_amdgcn_s_wait_asynccnt)
    __builtin_amdgcn_s_wait_asynccnt(0);
#else
    asm volatile("s_wait_asynccnt 0x0" ::: "memory");
#endif
}
#endif

// ---------------------------------------------------------------------------
// mass[b,a,h] = log_cosh( key[b,a,:] . mass_w[h,:] )
// ---------------------------------------------------------------------------
__global__ __launch_bounds__(256) void mass_kernel(
    const float* __restrict__ key, const float* __restrict__ mass_w,
    float* __restrict__ mass)
{
    int idx = blockIdx.x * blockDim.x + threadIdx.x;   // over B*A*H
    if (idx >= BATCH * SEQ * HEADS) return;
    int h  = idx % HEADS;
    int ba = idx / HEADS;
    const float* kr = key    + (size_t)ba * EMB;
    const float* wr = mass_w + (size_t)h  * EMB;
    float acc = 0.0f;
    #pragma unroll 4
    for (int e = 0; e < EMB; e += 4) {
        float4 a = *(const float4*)(kr + e);
        float4 b = *(const float4*)(wr + e);
        acc += a.x * b.x + a.y * b.y + a.z * b.z + a.w * b.w;
    }
    float ax = fabsf(acc);
    // log(cosh(x)) = |x| + log1p(exp(-2|x|)) - log(2)
    mass[idx] = ax + log1pf(__expf(-2.0f * ax)) - 0.69314718055994531f;
}

// ---------------------------------------------------------------------------
// One-wave 16x64 output tile of  OUT = X @ W^T   (K = EMB, all pitches EMB).
// One A fragment (X) is reused across four B fragments (rows of W), so X
// traffic is amortized 4x. fp32 loads -> f16 fragments on the fly.
// A-frag (16x32 f16): lane L holds row M=L&15; VGPR v holds K pair
//   k = (L>>4)*16 + 2v
// B-frag (32x16 f16): lane L holds col N=L&15; VGPR v holds K pair
//   k = (L>>4)*16 + 2v  -> B[k][n] = W[n][k]: contiguous reads of W rows.
// ---------------------------------------------------------------------------
template <bool HALF_OUT>
__device__ __forceinline__ void gemm_tile4(
    const float* __restrict__ X, const float* __restrict__ W,
    void* __restrict__ OUT, int mBase, int nBase)
{
    const int lane = threadIdx.x & 31;
    const int lo = lane & 15, hi = lane >> 4;
    v8f acc[4];
    #pragma unroll
    for (int j = 0; j < 4; ++j) acc[j] = (v8f)(0.0f);

    const float* xr = X + (size_t)(mBase + lo) * EMB;

    #pragma unroll
    for (int kc = 0; kc < EMB; kc += 32) {
        F16x16 A;
        #pragma unroll
        for (int v = 0; v < 8; ++v) {
            int ka = hi * 16 + 2 * v;
            float2 f = *(const float2*)(xr + kc + ka);
            hf2 fh = { (_Float16)f.x, (_Float16)f.y };
            A.u[v] = __builtin_bit_cast(uint32_t, fh);
        }
        #pragma unroll
        for (int j = 0; j < 4; ++j) {
            const float* wr = W + (size_t)(nBase + j * 16 + lo) * EMB + kc;
            F16x16 Bm;
            #pragma unroll
            for (int v = 0; v < 8; ++v) {
                float2 g = *(const float2*)(wr + hi * 16 + 2 * v);
                hf2 gh = { (_Float16)g.x, (_Float16)g.y };
                Bm.u[v] = __builtin_bit_cast(uint32_t, gh);
            }
            acc[j] = __builtin_amdgcn_wmma_f32_16x16x32_f16(
                false, A.v, false, Bm.v, (short)0, acc[j], false, false);
        }
    }

    #pragma unroll
    for (int j = 0; j < 4; ++j)
        #pragma unroll
        for (int r = 0; r < 8; ++r) {
            int m = mBase + r + 8 * hi;   // C/D layout: VGPR r -> row r (+8 hi)
            int n = nBase + j * 16 + lo;
            if constexpr (HALF_OUT)
                ((_Float16*)OUT)[(size_t)m * EMB + n] = (_Float16)acc[j][r];
            else
                ((float*)OUT)[(size_t)m * EMB + n] = acc[j][r];
        }
}

// grid = (EMB/64, M/16, 3): z selects {query->q16, key->k16, value->v16}
__global__ __launch_bounds__(32) void qkv_proj_kernel(
    const float* __restrict__ q, const float* __restrict__ k,
    const float* __restrict__ v, const float* __restrict__ in_proj_w,
    _Float16* __restrict__ q16, _Float16* __restrict__ k16,
    _Float16* __restrict__ v16)
{
    int z = blockIdx.z;
    const float* X = (z == 0) ? q : (z == 1) ? k : v;
    const float* W = in_proj_w + (size_t)z * EMB * EMB;
    _Float16*   O  = (z == 0) ? q16 : (z == 1) ? k16 : v16;
    gemm_tile4<true>(X, W, O, blockIdx.y * 16, blockIdx.x * 64);
}

__global__ __launch_bounds__(32) void oproj_kernel(
    const float* __restrict__ X, const float* __restrict__ W,
    float* __restrict__ O)
{
    gemm_tile4<false>(X, W, O, blockIdx.y * 16, blockIdx.x * 64);
}

// ---------------------------------------------------------------------------
// Flash attention with distance-decay bias.
// grid = (SEQ/64, BATCH*HEADS), block = 128 (4 waves).
// Block shares 64x64 K/V f16 tiles in LDS (async-filled when available);
// each wave owns 16 query rows, runs online softmax, and accumulates O in
// four 16x16 f32 fragments. 16 v_wmma per 64-key iteration.
// ---------------------------------------------------------------------------
__global__ __launch_bounds__(128) void attn_kernel(
    const _Float16* __restrict__ q16, const _Float16* __restrict__ k16,
    const _Float16* __restrict__ v16, const float* __restrict__ mass,
    const float* __restrict__ dist, float* __restrict__ o32)
{
    __shared__ __align__(16) _Float16 sK[ATILE][DHEAD];
    __shared__ __align__(16) _Float16 sV[ATILE][DHEAD];
    __shared__ float sMass[ATILE];
    __shared__ __align__(8) _Float16 sP[4][16][ATILE];   // per-wave P staging

    const int tid  = threadIdx.x;
    const int wave = tid >> 5, lane = tid & 31;
    const int lo = lane & 15, hi = lane >> 4;
    const int b = blockIdx.y >> 2, h = blockIdx.y & 3;
    const int qBase = blockIdx.x * 64 + wave * 16;

    const _Float16* qp = q16 + ((size_t)b * SEQ + qBase) * EMB + h * DHEAD;
    const _Float16* kp = k16 + (size_t)b * SEQ * EMB + h * DHEAD;
    const _Float16* vp = v16 + (size_t)b * SEQ * EMB + h * DHEAD;
    const float*    mp = mass + (size_t)b * SEQ * HEADS + h;
    const float*    dp = dist + ((size_t)b * SEQ + qBase) * (size_t)SEQ;

    // Q fragments: two D-chunks of 32 (D=64), resident for the whole pass.
    F16x16 aq[2];
    #pragma unroll
    for (int c = 0; c < 2; ++c)
        #pragma unroll
        for (int v = 0; v < 8; ++v) {
            int ka = hi * 16 + 2 * v;
            aq[c].u[v] = *(const uint32_t*)(qp + (size_t)lo * EMB + c * 32 + ka);
        }

    v8f oacc[4];
    float mrow[8], lrow[8];
    #pragma unroll
    for (int c = 0; c < 4; ++c) oacc[c] = (v8f)(0.0f);
    #pragma unroll
    for (int r = 0; r < 8; ++r) { mrow[r] = -1e30f; lrow[r] = 0.0f; }

    for (int a0 = 0; a0 < SEQ; a0 += ATILE) {
        // Fill K/V tiles: 64 rows x 64 halves = 8KB each = 512 x 16B.
#if HAVE_ASYNC_LDS
        #pragma unroll
        for (int it = 0; it < 4; ++it) {
            int idx = tid + it * 128;
            int row = idx >> 3, c16 = idx & 7;
            async_copy16(kp + (size_t)(a0 + row) * EMB + c16 * 8, &sK[row][c16 * 8]);
            async_copy16(vp + (size_t)(a0 + row) * EMB + c16 * 8, &sV[row][c16 * 8]);
        }
#else
        #pragma unroll
        for (int it = 0; it < 4; ++it) {
            int idx = tid + it * 128;
            int row = idx >> 3, c16 = idx & 7;
            *(uint4*)&sK[row][c16 * 8] =
                *(const uint4*)(kp + (size_t)(a0 + row) * EMB + c16 * 8);
            *(uint4*)&sV[row][c16 * 8] =
                *(const uint4*)(vp + (size_t)(a0 + row) * EMB + c16 * 8);
        }
#endif
        if (tid < ATILE) sMass[tid] = mp[(size_t)(a0 + tid) * HEADS];
#if HAVE_ASYNC_LDS
        wait_async_zero();   // own async fills landed in LDS before barrier
#endif
        __syncthreads();

        if (a0 + ATILE < SEQ) {   // global_prefetch_b8 for next tiles
            const _Float16* base = (tid < 64) ? kp : vp;
            __builtin_prefetch(base + (size_t)(a0 + ATILE + (tid & 63)) * EMB, 0, 1);
        }

        // S = Q K^T * scale - mass[a]*dist[i,a]; four 16-key sub-tiles.
        float ps[4][8];
        #pragma unroll
        for (int sub = 0; sub < 4; ++sub) {
            v8f c = (v8f)(0.0f);
            #pragma unroll
            for (int dc = 0; dc < 2; ++dc) {
                F16x16 bm;   // B[k][n] = K[sub*16+n][dc*32+k]: row reads of K tile
                #pragma unroll
                for (int v = 0; v < 8; ++v) {
                    int d = dc * 32 + hi * 16 + 2 * v;
                    bm.u[v] = *(const uint32_t*)&sK[sub * 16 + lo][d];
                }
                c = __builtin_amdgcn_wmma_f32_16x16x32_f16(
                    false, aq[dc].v, false, bm.v, (short)0, c, false, false);
            }
            float mw = sMass[sub * 16 + lo];
            #pragma unroll
            for (int r = 0; r < 8; ++r) {
                float dv = dp[(size_t)(r + 8 * hi) * SEQ + a0 + sub * 16 + lo];
                ps[sub][r] = c[r] * 0.125f - mw * dv;   // scale = 1/sqrt(64)
            }
        }

        // Online softmax; row stats per C-fragment register, reduced across
        // the 16-lane half-groups with shfl_xor (wave32).
        #pragma unroll
        for (int r = 0; r < 8; ++r) {
            float mx = fmaxf(fmaxf(ps[0][r], ps[1][r]), fmaxf(ps[2][r], ps[3][r]));
            #pragma unroll
            for (int msk = 1; msk < 16; msk <<= 1)
                mx = fmaxf(mx, __shfl_xor(mx, msk, 32));
            float mnew  = fmaxf(mrow[r], mx);
            float alpha = __expf(mrow[r] - mnew);
            mrow[r] = mnew;
            float rs = 0.0f;
            #pragma unroll
            for (int sub = 0; sub < 4; ++sub) {
                float p = __expf(ps[sub][r] - mnew);
                ps[sub][r] = p;
                rs += p;
            }
            #pragma unroll
            for (int msk = 1; msk < 16; msk <<= 1)
                rs += __shfl_xor(rs, msk, 32);
            lrow[r] = lrow[r] * alpha + rs;
            #pragma unroll
            for (int cc = 0; cc < 4; ++cc) oacc[cc][r] *= alpha;
        }

        // C-layout P -> LDS -> A-layout fragments (cross-lane transpose via
        // wave-private LDS region; same-wave DS ordering + explicit wait).
        #pragma unroll
        for (int sub = 0; sub < 4; ++sub)
            #pragma unroll
            for (int r = 0; r < 8; ++r)
                sP[wave][r + 8 * hi][sub * 16 + lo] = (_Float16)ps[sub][r];
        asm volatile("s_wait_dscnt 0" ::: "memory");

        F16x16 pa[2];
        #pragma unroll
        for (int kc = 0; kc < 2; ++kc)
            #pragma unroll
            for (int v = 0; v < 8; ++v) {
                int ka = hi * 16 + 2 * v;
                pa[kc].u[v] = *(const uint32_t*)&sP[wave][lo][kc * 32 + ka];
            }

        // O += P @ V over four 16-wide D chunks, K split in two 32-chunks.
        #pragma unroll
        for (int cc = 0; cc < 4; ++cc)
            #pragma unroll
            for (int kc = 0; kc < 2; ++kc) {
                F16x16 bv;   // B[k][n] = V[kc*32+k][cc*16+n]
                #pragma unroll
                for (int v = 0; v < 8; ++v) {
                    int k = kc * 32 + hi * 16 + 2 * v;
                    hf2 t = { sV[k][cc * 16 + lo], sV[k + 1][cc * 16 + lo] };
                    bv.u[v] = __builtin_bit_cast(uint32_t, t);
                }
                oacc[cc] = __builtin_amdgcn_wmma_f32_16x16x32_f16(
                    false, pa[kc].v, false, bv.v, (short)0, oacc[cc], false, false);
            }
        __syncthreads();
    }

    // Normalize and write O (B, I, H*D) fp32 for the output projection.
    #pragma unroll
    for (int r = 0; r < 8; ++r) {
        float inv = 1.0f / lrow[r];
        int i = qBase + r + 8 * hi;
        #pragma unroll
        for (int cc = 0; cc < 4; ++cc)
            o32[((size_t)b * SEQ + i) * EMB + h * DHEAD + cc * 16 + lo] =
                oacc[cc][r] * inv;
    }
}

// ---------------------------------------------------------------------------
extern "C" void kernel_launch(void* const* d_in, const int* in_sizes, int n_in,
                              void* d_out, int out_size, void* d_ws, size_t ws_size,
                              hipStream_t stream)
{
    const float* query      = (const float*)d_in[0];
    const float* key        = (const float*)d_in[1];
    const float* value      = (const float*)d_in[2];
    const float* distances  = (const float*)d_in[3];
    const float* in_proj_w  = (const float*)d_in[4];
    const float* out_proj_w = (const float*)d_in[5];
    const float* mass_w     = (const float*)d_in[6];
    float* out = (float*)d_out;

    const size_t M = (size_t)BATCH * SEQ;          // 8192 rows
    _Float16* q16  = (_Float16*)d_ws;
    _Float16* k16  = q16 + M * EMB;
    _Float16* v16  = k16 + M * EMB;
    float*    msb  = (float*)(v16 + M * EMB);      // mass (B, A, H)
    float*    o32  = msb + M * HEADS;              // attention output (B, I, E)

    mass_kernel<<<(BATCH * SEQ * HEADS + 255) / 256, 256, 0, stream>>>(
        key, mass_w, msb);

    qkv_proj_kernel<<<dim3(EMB / 64, M / 16, 3), 32, 0, stream>>>(
        query, key, value, in_proj_w, q16, k16, v16);

    attn_kernel<<<dim3(SEQ / 64, BATCH * HEADS), 128, 0, stream>>>(
        q16, k16, v16, msb, distances, o32);

    oproj_kernel<<<dim3(EMB / 64, M / 16, 1), 32, 0, stream>>>(
        o32, out_proj_w, out);
}
